// CrossAttnMem_66932770340952
// MI455X (gfx1250) — matrix operand order repaired and checked by reading, not compiled, not run on hardware
//
#include <hip/hip_runtime.h>

// CrossAttnMem on MI455X (gfx1250): bf16 WMMA pipeline with async LDS copies.
// Shapes: B=8, N=1024, C=128, CH=1024, D=B*CH=8192.
// Pipeline:
//   convert f32 -> bf16 (emb, Wq, Wk, Wv, Wo)
//   GEMM qkv:   q = emb_s @ Wq^T (bf16, M=8192,N=1024,K=128)
//               k,v scattered into k_flat/v_flat layout (N x B*CH)
//   GEMM score: scores[b] = q[b]^T @ k_flat  (M=1024,N=8192,K=1024, per batch)
//               + fused sum/sumsq atomics per batch -> stats
//   softmax:    x=(s-mean)*rstd, softmax over d=8192, in-place bf16
//   GEMM ctx:   ctx[b] = v_flat @ attn[b]^T  (M=1024,N=1024,K=8192)
//   GEMM out:   out = ctx @ Wo^T (f32 out,   M=8192,N=128, K=1024)
// GEMM core: 128 threads (4 wave32), 128x128 block tile, 64x64 per wave,
// K-step 32, v_wmma_f32_16x16x32_bf16, double-buffered LDS filled with
// global_load_async_to_lds_b128 (ASYNCcnt) for global->LDS overlap.

typedef unsigned short u16;
typedef unsigned int   u32;
typedef __attribute__((ext_vector_type(16))) __bf16 v16bf;
typedef __attribute__((ext_vector_type(8)))  float  v8f;

#define LN_EPS 1e-5f

__device__ __forceinline__ u16 f2bf(float f) {
  u32 u = __float_as_uint(f);
  u += 0x7FFFu + ((u >> 16) & 1u);          // round-to-nearest-even
  return (u16)(u >> 16);
}
__device__ __forceinline__ float bf2f(u16 h) {
  return __uint_as_float(((u32)h) << 16);
}

union FragU { v16bf v; u16 h[16]; u32 w[8]; };
union AccU  { v8f v;  float f[8]; };

// LDS byte address = low 32 bits of the generic pointer;
// ptrtoint also escapes the shared array so the
// compiler keeps the asm-written LDS live.
__device__ __forceinline__ u32 lds_addr(const u16* p) {
  return (u32)(size_t)(const void*)p;
}

__device__ __forceinline__ void async_b128(u32 ldsOff, u32 gOffBytes, const u16* base) {
  asm volatile("global_load_async_to_lds_b128 %0, %1, %2"
               :: "v"(ldsOff), "v"(gOffBytes), "s"(base) : "memory");
}
__device__ __forceinline__ void wait_async0() {
  asm volatile("s_wait_asynccnt 0" ::: "memory");
}

// ---------------------------------------------------------------------------
// Generic bf16 WMMA GEMM: C[m,n] = sum_k A[m,k]*B[k,n]
//  TA: A stored k-major (A[k*lda+m]) else m-major (A[m*lda+k])
//  TB: B stored n-major (B[n*ldb+k]) else k-major (B[k*ldb+n])
//  OUT_MODE: 0 = bf16 row-major (+batch stride sC)
//            1 = bf16 scatter C[(m&1023)*8192 + (m>>10)*1024 + n]  (k_flat/v_flat)
//            2 = f32 row-major
//  STATS: accumulate per-batch sum / sumsq of C into stats[2*bz(+1)]
// ---------------------------------------------------------------------------
template<bool TA, bool TB, int OUT_MODE, bool STATS>
__global__ __launch_bounds__(128)
void gemm_bf16_wmma(const u16* __restrict__ A, const u16* __restrict__ B,
                    void* __restrict__ Cv, float* __restrict__ stats,
                    int M, int N, int K, int lda, int ldb, int ldc,
                    long sA, long sB, long sC)
{
  constexpr int TM = 128, TN = 128, TK = 32;
  constexpr int PAD = 8;                       // keep 16B alignment for b128
  constexpr int ASZ = TA ? TK * (TM + PAD) : TM * (TK + PAD);
  constexpr int BSZ = TB ? TN * (TK + PAD) : TK * (TN + PAD);
  __shared__ u16 As[2][ASZ];
  __shared__ u16 Bs[2][BSZ];

  const int bz = blockIdx.z;
  A += (long)bz * sA;
  B += (long)bz * sB;
  const int m0 = blockIdx.y * TM;
  const int n0 = blockIdx.x * TN;
  const int tid  = threadIdx.x;
  const int lane = tid & 31;
  const int wave = tid >> 5;
  const int wm = (wave >> 1) * 64;             // 2x2 waves of 64x64
  const int wn = (wave & 1) * 64;

  v8f acc[4][4];
  #pragma unroll
  for (int i = 0; i < 4; ++i)
    #pragma unroll
    for (int j = 0; j < 4; ++j) { v8f z = {}; acc[i][j] = z; }

  const int nlo  = lane & 15;
  const int half = lane >> 4;

  // Stage one 128x32 (or 32x128) K-tile into LDS buffer via async copies.
  // 512 x 16B chunks per tile, 4 chunks per thread.
  auto stageA = [&](u16* dst, int k0) {
    #pragma unroll
    for (int it = 0; it < 4; ++it) {
      int c = it * 128 + tid;
      if (!TA) {                               // rows m, contiguous k
        int row = c >> 2, part = c & 3;
        u32 goff = (u32)(((long)(m0 + row) * lda + k0 + part * 8) * 2);
        async_b128(lds_addr(dst + row * (TK + PAD) + part * 8), goff, A);
      } else {                                 // rows k, contiguous m
        int row = c >> 4, part = c & 15;
        u32 goff = (u32)(((long)(k0 + row) * lda + m0 + part * 8) * 2);
        async_b128(lds_addr(dst + row * (TM + PAD) + part * 8), goff, A);
      }
    }
  };
  auto stageB = [&](u16* dst, int k0) {
    #pragma unroll
    for (int it = 0; it < 4; ++it) {
      int c = it * 128 + tid;
      if (!TB) {                               // rows k, contiguous n
        int row = c >> 4, part = c & 15;
        u32 goff = (u32)(((long)(k0 + row) * ldb + n0 + part * 8) * 2);
        async_b128(lds_addr(dst + row * (TN + PAD) + part * 8), goff, B);
      } else {                                 // rows n, contiguous k
        int row = c >> 2, part = c & 3;
        u32 goff = (u32)(((long)(n0 + row) * ldb + k0 + part * 8) * 2);
        async_b128(lds_addr(dst + row * (TK + PAD) + part * 8), goff, B);
      }
    }
  };

  // prologue: fill buffer 0
  stageA(As[0], 0);
  stageB(Bs[0], 0);
  wait_async0();
  __syncthreads();

  int buf = 0;
  for (int k0 = 0; k0 < K; k0 += TK) {
    const bool more = (k0 + TK) < K;
    if (more) {                                // issue next tile, no wait yet
      stageA(As[buf ^ 1], k0 + TK);
      stageB(Bs[buf ^ 1], k0 + TK);
    }
    const u16* Ac = As[buf];
    const u16* Bc = Bs[buf];

    // ---- load fragments per CDNA5 WMMA VGPR layouts ----
    FragU af[4], bfr[4];
    #pragma unroll
    for (int i = 0; i < 4; ++i) {
      int mb = wm + i * 16 + nlo;              // A row M = lane&15
      int ho = half ? 8 : 0;
      #pragma unroll
      for (int p = 0; p < 8; ++p) {
        int kk = ((p < 4) ? (2 * p) : (16 + 2 * (p - 4))) + ho;
        if (!TA) af[i].w[p] = *(const u32*)(Ac + mb * (TK + PAD) + kk);
        else {
          af[i].h[2 * p]     = Ac[kk       * (TM + PAD) + mb];
          af[i].h[2 * p + 1] = Ac[(kk + 1) * (TM + PAD) + mb];
        }
      }
    }
    #pragma unroll
    for (int j = 0; j < 4; ++j) {
      int nb = wn + j * 16 + nlo;              // B col N = lane&15
      int ho = half ? 8 : 0;
      #pragma unroll
      for (int p = 0; p < 8; ++p) {
        int kk = ((p < 4) ? (2 * p) : (16 + 2 * (p - 4))) + ho;
        if (!TB) {
          bfr[j].h[2 * p]     = Bc[kk       * (TN + PAD) + nb];
          bfr[j].h[2 * p + 1] = Bc[(kk + 1) * (TN + PAD) + nb];
        } else bfr[j].w[p] = *(const u32*)(Bc + nb * (TK + PAD) + kk);
      }
    }
    // ---- 16 WMMA ops per K-step ----
    #pragma unroll
    for (int i = 0; i < 4; ++i)
      #pragma unroll
      for (int j = 0; j < 4; ++j)
        acc[i][j] = __builtin_amdgcn_wmma_f32_16x16x32_bf16(
            false, af[i].v, false, bfr[j].v, (short)0, acc[i][j], false, false);

    if (more) wait_async0();                   // next buffer landed
    __syncthreads();                           // everyone done with this buffer
    buf ^= 1;
  }

  // ---- epilogue: D layout VGPR r -> M = r + 8*half, N = lane&15 ----
  float ssum = 0.f, ssq = 0.f;
  const int mh = half ? 8 : 0;
  #pragma unroll
  for (int i = 0; i < 4; ++i) {
    #pragma unroll
    for (int j = 0; j < 4; ++j) {
      AccU cu; cu.v = acc[i][j];
      #pragma unroll
      for (int r = 0; r < 8; ++r) {
        int m = m0 + wm + i * 16 + mh + r;
        int n = n0 + wn + j * 16 + nlo;
        float val = cu.f[r];
        if (STATS) { ssum += val; ssq += val * val; }
        if (OUT_MODE == 0) {
          u16* C = (u16*)Cv + (long)bz * sC;
          C[(long)m * ldc + n] = f2bf(val);
        } else if (OUT_MODE == 1) {
          u16* C = (u16*)Cv;   // k_flat[n_tok, b*1024 + ch]
          C[(long)(m & 1023) * 8192 + (long)(m >> 10) * 1024 + n] = f2bf(val);
        } else {
          float* C = (float*)Cv + (long)bz * sC;
          C[(long)m * ldc + n] = val;
        }
      }
    }
  }
  if (STATS) {
    #pragma unroll
    for (int o = 16; o >= 1; o >>= 1) {
      ssum += __shfl_xor(ssum, o, 32);
      ssq  += __shfl_xor(ssq,  o, 32);
    }
    if (lane == 0) {
      atomicAdd(&stats[2 * bz],     ssum);
      atomicAdd(&stats[2 * bz + 1], ssq);
    }
  }
}

// ---------------------------------------------------------------------------
// In-place normalize + softmax over d=8192 per (b, c') row, bf16 in/out.
// ---------------------------------------------------------------------------
__global__ __launch_bounds__(256)
void softmax_norm(u16* __restrict__ S, const float* __restrict__ stats)
{
  const long row = blockIdx.x;          // b*1024 + c'
  const int  b   = (int)(row >> 10);
  u16* p = S + row * 8192;
  const int tid = threadIdx.x;

  const float CNT  = 8388608.0f;        // 1024*8192
  const float mean = stats[2 * b] / CNT;
  const float var  = stats[2 * b + 1] / CNT - mean * mean;
  const float rstd = rsqrtf(var + LN_EPS);

  float vals[32];
  float lmax = -3.4e38f;
  #pragma unroll
  for (int i = 0; i < 32; ++i) {
    float x = (bf2f(p[tid + i * 256]) - mean) * rstd;
    vals[i] = x;
    lmax = fmaxf(lmax, x);
  }
  __shared__ float red[8];
  #pragma unroll
  for (int o = 16; o >= 1; o >>= 1) lmax = fmaxf(lmax, __shfl_xor(lmax, o, 32));
  if ((tid & 31) == 0) red[tid >> 5] = lmax;
  __syncthreads();
  float bmax = red[0];
  #pragma unroll
  for (int w = 1; w < 8; ++w) bmax = fmaxf(bmax, red[w]);
  __syncthreads();

  float lsum = 0.f;
  #pragma unroll
  for (int i = 0; i < 32; ++i) {
    float e = __expf(vals[i] - bmax);
    vals[i] = e;
    lsum += e;
  }
  #pragma unroll
  for (int o = 16; o >= 1; o >>= 1) lsum += __shfl_xor(lsum, o, 32);
  if ((tid & 31) == 0) red[tid >> 5] = lsum;
  __syncthreads();
  float bsum = 0.f;
  #pragma unroll
  for (int w = 0; w < 8; ++w) bsum += red[w];
  const float inv = 1.0f / bsum;
  #pragma unroll
  for (int i = 0; i < 32; ++i) p[tid + i * 256] = f2bf(vals[i] * inv);
}

// ---------------------------------------------------------------------------
__global__ void f32_to_bf16_k(const float* __restrict__ src, u16* __restrict__ dst, long n)
{
  long i = (long)blockIdx.x * blockDim.x + threadIdx.x;
  long stride = (long)gridDim.x * blockDim.x;
  for (; i < n; i += stride) dst[i] = f2bf(src[i]);
}

__global__ void zero_stats_k(float* __restrict__ s)
{
  if (threadIdx.x < 16) s[threadIdx.x] = 0.f;
}

// ---------------------------------------------------------------------------
extern "C" void kernel_launch(void* const* d_in, const int* in_sizes, int n_in,
                              void* d_out, int out_size, void* d_ws, size_t ws_size,
                              hipStream_t stream)
{
  (void)in_sizes; (void)n_in; (void)out_size; (void)ws_size;
  const float* emb = (const float*)d_in[0];   // (16,1024,128)
  const float* Wq  = (const float*)d_in[1];   // (1024,128)
  const float* Wk  = (const float*)d_in[2];
  const float* Wv  = (const float*)d_in[3];
  const float* Wo  = (const float*)d_in[4];   // (128,1024)
  float* out = (float*)d_out;                 // (8,1024,128)

  // workspace partition (bytes)
  char* ws = (char*)d_ws;
  u16* embB = (u16*)ws; ws += (size_t)2097152 * 2;   // 16*1024*128
  u16* WqB  = (u16*)ws; ws += (size_t)131072 * 2;    // 1024*128
  u16* WkB  = (u16*)ws; ws += (size_t)131072 * 2;
  u16* WvB  = (u16*)ws; ws += (size_t)131072 * 2;
  u16* WoB  = (u16*)ws; ws += (size_t)131072 * 2;    // 128*1024
  u16* qB   = (u16*)ws; ws += (size_t)8388608 * 2;   // 8*1024 x 1024
  u16* kB   = (u16*)ws; ws += (size_t)8388608 * 2;   // 1024 x 8192
  u16* vB   = (u16*)ws; ws += (size_t)8388608 * 2;   // 1024 x 8192
  u16* scB  = (u16*)ws; ws += (size_t)67108864 * 2;  // 8 x 1024 x 8192 (scores/attn)
  u16* ctxB = (u16*)ws; ws += (size_t)8388608 * 2;   // 8 x 1024 x 1024
  float* stats = (float*)ws;                         // 16 floats

  zero_stats_k<<<1, 32, 0, stream>>>(stats);

  f32_to_bf16_k<<<2048, 256, 0, stream>>>(emb, embB, 2097152L);
  f32_to_bf16_k<<<512,  256, 0, stream>>>(Wq,  WqB,  131072L);
  f32_to_bf16_k<<<512,  256, 0, stream>>>(Wk,  WkB,  131072L);
  f32_to_bf16_k<<<512,  256, 0, stream>>>(Wv,  WvB,  131072L);
  f32_to_bf16_k<<<512,  256, 0, stream>>>(Wo,  WoB,  131072L);

  // q = emb_s @ Wq^T  : M=8192, N=1024, K=128
  gemm_bf16_wmma<false, true, 0, false><<<dim3(8, 64, 1), 128, 0, stream>>>(
      embB, WqB, qB, nullptr, 8192, 1024, 128, 128, 128, 1024, 0, 0, 0);
  // k_flat / v_flat from emb_t, scattered layout (N x 8192)
  gemm_bf16_wmma<false, true, 1, false><<<dim3(8, 64, 1), 128, 0, stream>>>(
      embB + (size_t)8192 * 128, WkB, kB, nullptr, 8192, 1024, 128, 128, 128, 0, 0, 0, 0);
  gemm_bf16_wmma<false, true, 1, false><<<dim3(8, 64, 1), 128, 0, stream>>>(
      embB + (size_t)8192 * 128, WvB, vB, nullptr, 8192, 1024, 128, 128, 128, 0, 0, 0, 0);

  // scores[b] = q[b]^T @ k_flat : M=1024, N=8192, K=1024 (+ stats)
  gemm_bf16_wmma<true, false, 0, true><<<dim3(64, 8, 8), 128, 0, stream>>>(
      qB, kB, scB, stats, 1024, 8192, 1024, 1024, 8192, 8192,
      1048576L, 0L, 8388608L);

  // layernorm + softmax in place over d
  softmax_norm<<<8192, 256, 0, stream>>>(scB, stats);

  // ctx[b] = v_flat @ attn[b]^T : M=1024 (tokens), N=1024 (ch), K=8192
  gemm_bf16_wmma<false, true, 0, false><<<dim3(8, 8, 8), 128, 0, stream>>>(
      vB, scB, ctxB, nullptr, 1024, 1024, 8192, 8192, 8192, 1024,
      0L, 8388608L, 1048576L);

  // out = ctx @ Wo^T : M=8192, N=128, K=1024 -> f32
  gemm_bf16_wmma<false, true, 2, false><<<dim3(1, 64, 1), 128, 0, stream>>>(
      ctxB, WoB, out, nullptr, 8192, 128, 1024, 1024, 1024, 128, 0, 0, 0);
}
